// highwayNet_d_8538394984638
// MI455X (gfx1250) — matrix operand order repaired and checked by reading, not compiled, not run on hardware
//
#include <hip/hip_runtime.h>

typedef _Float16 half16 __attribute__((ext_vector_type(16)));
typedef float    floatx8 __attribute__((ext_vector_type(8)));

#define ZACC8 {0.f,0.f,0.f,0.f,0.f,0.f,0.f,0.f}

__device__ __forceinline__ float leakyf(float x) { return x > 0.0f ? x : 0.1f * x; }
__device__ __forceinline__ float sigmoidf_(float x) { return 1.0f / (1.0f + __expf(-x)); }

// D = A(16x32 f16) * B(32x16 f16) + C(16x16 f32)
__device__ __forceinline__ floatx8 wmma16x16x32(half16 a, half16 b, floatx8 c) {
  return __builtin_amdgcn_wmma_f32_16x16x32_f16(false, a, false, b, (short)0, c, false, false);
}

// A-fragment K order chosen here (contiguous, B staged to match): lanes 0-15 -> K in {0..15},
// lanes 16-31 -> K in {16..31}.  KMAP(i) with lo8 = (lane>=16)*8.
#define KMAP(i) (2 * lo8 + (i))

// ---------------------------------------------------------------------------
// Kernel 1: bi-directional GRU encoder + seq_enc  (one wave, 16 vehicles/block)
// ---------------------------------------------------------------------------
__global__ __launch_bounds__(32) void gru_kernel(
    const float* __restrict__ scene,
    const float* __restrict__ W_emb, const float* __restrict__ b_emb,
    const float* __restrict__ Wih_f, const float* __restrict__ Whh_f,
    const float* __restrict__ bih_f, const float* __restrict__ bhh_f,
    const float* __restrict__ Wih_b, const float* __restrict__ Whh_b,
    const float* __restrict__ bih_b, const float* __restrict__ bhh_b,
    const float* __restrict__ Wm, const float* __restrict__ bm,
    float* __restrict__ out, int N)
{
  // f16 B-fragments: frag f occupies [f*512, f*512+512) halves, lane L owns 16 halves at L*16
  __shared__ __align__(32) _Float16 WBih[12 * 512];   // 12 gate-col blocks, K=32
  __shared__ __align__(32) _Float16 WBhh[24 * 512];   // 12 blocks x 2 K-tiles of 32
  __shared__ float    sc[16 * 32];                    // scene tile [v][j]
  __shared__ float    hlds[16 * 64];                  // current hidden state (fp32)
  __shared__ __align__(32) _Float16 hfl[16 * 64];     // forward final hidden (f16)
  __shared__ float    gbuf[16 * 192];                 // gates: r,z fused; n-gate x part
  __shared__ float    hbuf[16 * 64];                  // n-gate h part
  __shared__ float    bihs[192];
  __shared__ float    bhhs[192];

  const int L = threadIdx.x;
  const int lane16 = L & 15;
  const int hi = L >> 4;
  const int lo8 = hi << 3;
  const int v0 = blockIdx.x * 16;

  __builtin_prefetch(Whh_f, 0, 0);
  __builtin_prefetch(Whh_b, 0, 0);

  // scene tile: 512 floats, 16 per lane
  #pragma unroll
  for (int j = 0; j < 16; ++j)
    sc[L * 16 + j] = scene[(size_t)v0 * 32 + L * 16 + j];

  // per-lane embedding weights on this lane's K-set
  float we0[16], we1[16], be[16];
  #pragma unroll
  for (int i = 0; i < 16; ++i) {
    int e = KMAP(i);
    we0[i] = W_emb[e];
    we1[i] = W_emb[32 + e];
    be[i]  = b_emb[e];
  }

  #pragma unroll 1
  for (int dir = 0; dir < 2; ++dir) {
    const float* Wih = dir ? Wih_b : Wih_f;   // (192,32) row-major
    const float* Whh = dir ? Whh_b : Whh_f;   // (192,64) row-major
    const float* bih = dir ? bih_b : bih_f;
    const float* bhh = dir ? bhh_b : bhh_f;

    // Stage B fragments: B[k][n] = W[col][k]; lane = column nb*16+lane16,
    // K-run = (hi*16 .. hi*16+15) within the K-tile (contiguous row read).
    #pragma unroll 1
    for (int f = 0; f < 12; ++f) {
      int col = f * 16 + lane16;
      int klo = hi * 16;
      #pragma unroll
      for (int i = 0; i < 16; ++i)
        WBih[f * 512 + L * 16 + i] = (_Float16)Wih[col * 32 + klo + i];
    }
    #pragma unroll 1
    for (int f = 0; f < 24; ++f) {
      int nb = f >> 1, kt = f & 1;
      int col = nb * 16 + lane16;
      int klo = kt * 32 + hi * 16;
      #pragma unroll
      for (int i = 0; i < 16; ++i)
        WBhh[f * 512 + L * 16 + i] = (_Float16)Whh[col * 64 + klo + i];
    }
    for (int i = L; i < 192; i += 32) { bihs[i] = bih[i]; bhhs[i] = bhh[i]; }
    for (int i = L; i < 1024; i += 32) hlds[i] = 0.0f;
    __syncthreads();

    #pragma unroll 1
    for (int s = 0; s < 16; ++s) {
      int t = dir ? (15 - s) : s;
      float x = sc[lane16 * 32 + t];
      float y = sc[lane16 * 32 + 16 + t];
      half16 aemb, ah0, ah1;
      #pragma unroll
      for (int i = 0; i < 16; ++i) {
        int k = KMAP(i);
        aemb[i] = (_Float16)leakyf(x * we0[i] + y * we1[i] + be[i]);
        ah0[i]  = (_Float16)hlds[lane16 * 64 + k];
        ah1[i]  = (_Float16)hlds[lane16 * 64 + 32 + k];
      }
      // r,z gate blocks: fused x+h accumulation
      #pragma unroll 1
      for (int nb = 0; nb < 8; ++nb) {
        floatx8 acc = ZACC8;
        acc = wmma16x16x32(aemb, *(const half16*)(WBih + nb * 512 + L * 16), acc);
        acc = wmma16x16x32(ah0,  *(const half16*)(WBhh + (nb * 2) * 512 + L * 16), acc);
        acc = wmma16x16x32(ah1,  *(const half16*)(WBhh + (nb * 2 + 1) * 512 + L * 16), acc);
        int col = nb * 16 + lane16;
        float bias = bihs[col] + bhhs[col];
        #pragma unroll
        for (int rr = 0; rr < 8; ++rr)
          gbuf[(rr + hi * 8) * 192 + col] = acc[rr] + bias;
      }
      // n-gate blocks: keep x and h parts separate (tanh(xn + r*hn))
      #pragma unroll 1
      for (int nb = 8; nb < 12; ++nb) {
        floatx8 accx = ZACC8, acch = ZACC8;
        accx = wmma16x16x32(aemb, *(const half16*)(WBih + nb * 512 + L * 16), accx);
        acch = wmma16x16x32(ah0,  *(const half16*)(WBhh + (nb * 2) * 512 + L * 16), acch);
        acch = wmma16x16x32(ah1,  *(const half16*)(WBhh + (nb * 2 + 1) * 512 + L * 16), acch);
        int col = nb * 16 + lane16;
        #pragma unroll
        for (int rr = 0; rr < 8; ++rr) {
          int v = rr + hi * 8;
          gbuf[v * 192 + col] = accx[rr] + bihs[col];
          hbuf[v * 64 + (col - 128)] = acch[rr] + bhhs[col];
        }
      }
      __syncthreads();
      // elementwise GRU update: each lane owns 32 (v,c) cells
      {
        int v = L >> 1;
        int cb = (L & 1) << 5;
        #pragma unroll 1
        for (int j = 0; j < 32; ++j) {
          int c = cb + j;
          float rg = sigmoidf_(gbuf[v * 192 + c]);
          float zg = sigmoidf_(gbuf[v * 192 + 64 + c]);
          float xn = gbuf[v * 192 + 128 + c];
          float hn = hbuf[v * 64 + c];
          float ng = tanhf(xn + rg * hn);
          float hp = hlds[v * 64 + c];
          hlds[v * 64 + c] = (1.0f - zg) * ng + zg * hp;
        }
      }
      __syncthreads();
    }

    if (dir == 0) {
      for (int i = L; i < 1024; i += 32) hfl[i] = (_Float16)hlds[i];
      __syncthreads();
    }
  }

  // hm = 0.5*(h_f + h_b); seq_enc = leaky(hm @ Wm + bm) -> full_enc[:, 0:32]
  for (int i = L; i < 1024; i += 32) hlds[i] = 0.5f * (hlds[i] + (float)hfl[i]);
  __syncthreads();

  half16 am0, am1;
  #pragma unroll
  for (int i = 0; i < 16; ++i) {
    int k = KMAP(i);
    am0[i] = (_Float16)hlds[lane16 * 64 + k];
    am1[i] = (_Float16)hlds[lane16 * 64 + 32 + k];
  }
  #pragma unroll 1
  for (int cbk = 0; cbk < 2; ++cbk) {
    int c = cbk * 16 + lane16;
    half16 b0, b1;
    #pragma unroll
    for (int i = 0; i < 16; ++i) {
      int k = hi * 16 + i;
      b0[i] = (_Float16)Wm[k * 32 + c];          // Wm is (64,32), used untransposed
      b1[i] = (_Float16)Wm[(32 + k) * 32 + c];
    }
    floatx8 acc = ZACC8;
    acc = wmma16x16x32(am0, b0, acc);
    acc = wmma16x16x32(am1, b1, acc);
    float bias = bm[c];
    #pragma unroll
    for (int rr = 0; rr < 8; ++rr) {
      int v = rr + hi * 8;
      out[(size_t)3 * N + (size_t)(v0 + v) * 64 + c] = leakyf(acc[rr] + bias);
    }
  }
}

// ---------------------------------------------------------------------------
// Kernel 2: pairwise spatial MLP + pooling + output head (one wave per (g,a))
// ---------------------------------------------------------------------------
__global__ __launch_bounds__(32) void spatial_kernel(
    const float* __restrict__ scene, const int* __restrict__ index_div,
    const float* __restrict__ Ws1, const float* __restrict__ bs1,
    const float* __restrict__ Ws2, const float* __restrict__ bs2,
    const float* __restrict__ Wo1, const float* __restrict__ bo1,
    const float* __restrict__ Wo2, const float* __restrict__ bo2,
    float* __restrict__ out, int N)
{
  __shared__ float ftile[32 * 32];   // group member features
  __shared__ float M1[32 * 64];      // leaky(rel @ Ws1 + bs1)
  __shared__ float fullv[64];
  __shared__ float t1s[16];

  const int r = blockIdx.x;          // row of pooled/full_enc == vehicle id
  const int g = r >> 5;
  const int a = r & 31;              // ego slot within group
  const int L = threadIdx.x;
  const int lane16 = L & 15;
  const int hi = L >> 4;
  const int lo8 = hi << 3;

  // gather group features via index_div (one row per lane)
  {
    int idxm = index_div[(g << 5) + L];
    const float* srow = scene + (size_t)idxm * 32;
    #pragma unroll
    for (int j = 0; j < 32; ++j) ftile[L * 32 + j] = srow[j];
  }
  __syncthreads();

  // ego-row features on this lane's K-set
  float fa[16];
  #pragma unroll
  for (int i = 0; i < 16; ++i) fa[i] = ftile[a * 32 + KMAP(i)];

  // GEMM1: rel (32x32) @ Ws1 (32x64)
  #pragma unroll 1
  for (int mt = 0; mt < 2; ++mt) {
    int brow = mt * 16 + lane16;
    half16 A;
    #pragma unroll
    for (int i = 0; i < 16; ++i)
      A[i] = (_Float16)(ftile[brow * 32 + KMAP(i)] - fa[i]);
    #pragma unroll 1
    for (int cb = 0; cb < 4; ++cb) {
      int c = cb * 16 + lane16;
      half16 B;
      #pragma unroll
      for (int i = 0; i < 16; ++i)
        B[i] = (_Float16)Ws1[(hi * 16 + i) * 64 + c];   // B[k][n] = Ws1[k][c]
      floatx8 acc = ZACC8;
      acc = wmma16x16x32(A, B, acc);
      float bias = bs1[c];
      #pragma unroll
      for (int rr = 0; rr < 8; ++rr)
        M1[(mt * 16 + rr + hi * 8) * 64 + c] = leakyf(acc[rr] + bias);
    }
  }
  __syncthreads();

  // GEMM2: M1 (32x64) @ Ws2 (64x32), fold leaky+bias and column-sum on the fly
  float colsum0 = 0.0f, colsum1 = 0.0f;
  #pragma unroll 1
  for (int mt = 0; mt < 2; ++mt) {
    int brow = mt * 16 + lane16;
    half16 A0, A1;
    #pragma unroll
    for (int i = 0; i < 16; ++i) {
      int k = KMAP(i);
      A0[i] = (_Float16)M1[brow * 64 + k];
      A1[i] = (_Float16)M1[brow * 64 + 32 + k];
    }
    #pragma unroll 1
    for (int cb = 0; cb < 2; ++cb) {
      int c = cb * 16 + lane16;
      half16 B0, B1;
      #pragma unroll
      for (int i = 0; i < 16; ++i) {
        int k = hi * 16 + i;
        B0[i] = (_Float16)Ws2[k * 32 + c];
        B1[i] = (_Float16)Ws2[(32 + k) * 32 + c];
      }
      floatx8 acc = ZACC8;
      acc = wmma16x16x32(A0, B0, acc);
      acc = wmma16x16x32(A1, B1, acc);
      float bias = bs2[c];
      float s = 0.0f;
      #pragma unroll
      for (int rr = 0; rr < 8; ++rr) s += leakyf(acc[rr] + bias);
      if (cb == 0) colsum0 += s; else colsum1 += s;
    }
  }
  // combine the two row-halves (lane L and L+16 hold the same column)
  colsum0 += __shfl_xor(colsum0, 16, 32);
  colsum1 += __shfl_xor(colsum1, 16, 32);

  // pooled = (colsum - zero_enc) / (NG-1); write full_enc[:, 32:64]
  #pragma unroll 1
  for (int cb = 0; cb < 2; ++cb) {
    int c = cb * 16 + lane16;
    float ze = bs2[c];
    #pragma unroll 1
    for (int k = 0; k < 64; ++k) ze += leakyf(bs1[k]) * Ws2[k * 32 + c];
    ze = leakyf(ze);
    float cs = cb ? colsum1 : colsum0;
    float pooled = (cs - ze) * (1.0f / 31.0f);
    if (hi == 0) {
      out[(size_t)3 * N + (size_t)r * 64 + 32 + c] = pooled;
      fullv[32 + c] = pooled;
    }
  }
  // seq_enc part from kernel 1 output
  fullv[L] = out[(size_t)3 * N + (size_t)r * 64 + L];
  __syncthreads();

  // output head: leaky(full @ Wo1 + bo1) @ Wo2 + bo2
  {
    int j = lane16;
    float t1 = bo1[j];
    #pragma unroll 1
    for (int k = 0; k < 64; ++k) t1 += fullv[k] * Wo1[k * 16 + j];
    t1 = leakyf(t1);
    if (hi == 0) t1s[j] = t1;
  }
  __syncthreads();
  if (L < 3) {
    float lg = bo2[L];
    #pragma unroll
    for (int j = 0; j < 16; ++j) lg += t1s[j] * Wo2[j * 3 + L];
    out[(size_t)r * 3 + L] = lg;
  }
}

// ---------------------------------------------------------------------------
extern "C" void kernel_launch(void* const* d_in, const int* in_sizes, int n_in,
                              void* d_out, int out_size, void* d_ws, size_t ws_size,
                              hipStream_t stream) {
  const float* scene     = (const float*)d_in[0];
  // d_in[1] condition, d_in[2] hero_pos_index, d_in[3] nbrs_pos_index: unused by reference
  const int*   index_div = (const int*)  d_in[4];
  const float* W_emb = (const float*)d_in[5];
  const float* b_emb = (const float*)d_in[6];
  const float* Wih_f = (const float*)d_in[7];
  const float* Whh_f = (const float*)d_in[8];
  const float* bih_f = (const float*)d_in[9];
  const float* bhh_f = (const float*)d_in[10];
  const float* Wih_b = (const float*)d_in[11];
  const float* Whh_b = (const float*)d_in[12];
  const float* bih_b = (const float*)d_in[13];
  const float* bhh_b = (const float*)d_in[14];
  const float* Wm  = (const float*)d_in[15];
  const float* bm  = (const float*)d_in[16];
  const float* Ws1 = (const float*)d_in[17];
  const float* bs1 = (const float*)d_in[18];
  const float* Ws2 = (const float*)d_in[19];
  const float* bs2 = (const float*)d_in[20];
  const float* Wo1 = (const float*)d_in[21];
  const float* bo1 = (const float*)d_in[22];
  const float* Wo2 = (const float*)d_in[23];
  const float* bo2 = (const float*)d_in[24];
  float* out = (float*)d_out;

  const int N = in_sizes[0] / 32;   // 32768

  gru_kernel<<<N / 16, 32, 0, stream>>>(scene, W_emb, b_emb,
      Wih_f, Whh_f, bih_f, bhh_f, Wih_b, Whh_b, bih_b, bhh_b, Wm, bm, out, N);
  spatial_kernel<<<N, 32, 0, stream>>>(scene, index_div, Ws1, bs1, Ws2, bs2,
      Wo1, bo1, Wo2, bo2, out, N);
}
